// CompoundGNN_82394652606592
// MI455X (gfx1250) — matrix operand order repaired and checked by reading, not compiled, not run on hardware
//
#include <hip/hip_runtime.h>
#include <hip/hip_bf16.h>

// ---------------- CDNA5 WMMA types ----------------
typedef __attribute__((ext_vector_type(16))) __bf16 v16bf;
typedef __attribute__((ext_vector_type(8)))  float  v8f;

__device__ __forceinline__ unsigned short f2bf(float f) {
    unsigned int u = __float_as_uint(f);
    unsigned int r = (u + 0x7FFFu + ((u >> 16) & 1u)) >> 16;
    return (unsigned short)r;
}

// ---------------- degree / norm ----------------
__global__ void k_init_deg(float* deg, int n) {
    int i = blockIdx.x * blockDim.x + threadIdx.x;
    if (i < n) deg[i] = 1.0f;   // self-loop contributes 1
}
__global__ void k_add_deg(const int* __restrict__ dst, float* deg, int e) {
    int i = blockIdx.x * blockDim.x + threadIdx.x;
    if (i < e) atomicAdd(&deg[dst[i]], 1.0f);
}
__global__ void k_dinv(float* deg, int n) {
    int i = blockIdx.x * blockDim.x + threadIdx.x;
    if (i < n) deg[i] = rsqrtf(deg[i]);   // deg >= 1 always
}

// ---------------- conversions / packing ----------------
__global__ void k_f32_to_bf16(const float* __restrict__ in, unsigned short* __restrict__ out, long long n4) {
    long long i = (long long)blockIdx.x * blockDim.x + threadIdx.x;
    if (i >= n4) return;
    float4 v = *(const float4*)(in + i * 4);
    ushort4 o;
    o.x = f2bf(v.x); o.y = f2bf(v.y); o.z = f2bf(v.z); o.w = f2bf(v.w);
    *(ushort4*)(out + i * 4) = o;
}

// Pack W [K x M] f32 row-major into bf16 B-fragments:
//   frag element i (0..15) of lane l for tile (kt, nt):
//     k = kt*32 + (i<8 ? 0 : 16) + (l<16 ? 0 : 8) + (i&7),  col = nt*16 + (l&15)
//   stored at ((nt*numKt + kt)*32 + l)*16 + i
__global__ void k_pack_w(const float* __restrict__ W, unsigned short* __restrict__ out, int K, int M) {
    int idx = blockIdx.x * blockDim.x + threadIdx.x;
    if (idx >= K * M) return;
    int i    = idx & 15;
    int l    = (idx >> 4) & 31;
    int tile = idx >> 9;
    int numKt = K >> 5;
    int kt = tile % numKt;
    int nt = tile / numKt;
    int k   = kt * 32 + ((i & 8) ? 16 : 0) + ((l & 16) ? 8 : 0) + (i & 7);
    int col = nt * 16 + (l & 15);
    out[idx] = f2bf(W[(size_t)k * M + col]);
}

// ---------------- WMMA GEMM: C[rows x M] = A[rows x K](bf16) @ Bpack + bias ----------------
// Block = 256 threads (8 waves). Block tile 128(M) x 128(N); wave w owns rows
// [mrow, mrow+16) and all 128 columns (8 WMMA accumulators).
// All 8 B-fragments of a k-step are loaded into independent registers before
// the 8 WMMAs issue, and the next k-step's A-fragment is prefetched, so VMEM
// latency overlaps the matrix pipe instead of serializing on loadcnt==0.
__global__ __launch_bounds__(256) void k_gemm_wmma(
    const unsigned short* __restrict__ A,   // bf16 row-major [rows x K]
    const unsigned short* __restrict__ Bp,  // packed fragments
    float* __restrict__ C,                  // [rows x M]
    const float* __restrict__ bias,         // may be null
    int K, int M, int fuse_relu)
{
    const int lane  = threadIdx.x & 31;
    const int wave  = threadIdx.x >> 5;
    const int mrow  = blockIdx.x * 128 + wave * 16;
    const int ncol0 = blockIdx.y * 128;
    const int numKt = K >> 5;

    const int row    = mrow + (lane & 15);
    const int kbaseA = (lane < 16) ? 0 : 8;
    const unsigned short* arow = A + (size_t)row * K;
    // Base of this wave's first B fragment (nt = ncol0/16), lane-resolved.
    const unsigned short* bbase = Bp + ((size_t)(ncol0 >> 4) * numKt * 32 + lane) * 16;
    const size_t bTileStride = (size_t)numKt * 512;   // elements between adjacent nt tiles
    const size_t bKtStride   = 512;                   // elements between adjacent kt tiles

    v8f acc[8];
#pragma unroll
    for (int j = 0; j < 8; ++j) acc[j] = (v8f){0.f,0.f,0.f,0.f,0.f,0.f,0.f,0.f};

    union Frag { v16bf v; uint4 q[2]; };

    Frag a;
    a.q[0] = *(const uint4*)(arow + kbaseA);
    a.q[1] = *(const uint4*)(arow + 16 + kbaseA);

    for (int kt = 0; kt < numKt; ++kt) {
        // Load all 8 B fragments for this k-step into independent registers.
        Frag b[8];
#pragma unroll
        for (int j = 0; j < 8; ++j) {
            const unsigned short* bp = bbase + (size_t)j * bTileStride + (size_t)kt * bKtStride;
            b[j].q[0] = *(const uint4*)(bp);
            b[j].q[1] = *(const uint4*)(bp + 8);
        }
        // Prefetch next k-step's A fragment.
        Frag a_next;
        if (kt + 1 < numKt) {
            a_next.q[0] = *(const uint4*)(arow + (kt + 1) * 32 + kbaseA);
            a_next.q[1] = *(const uint4*)(arow + (kt + 1) * 32 + 16 + kbaseA);
        }
        // 8 back-to-back WMMAs on independent accumulators.
#pragma unroll
        for (int j = 0; j < 8; ++j) {
            acc[j] = __builtin_amdgcn_wmma_f32_16x16x32_bf16(
                false, a.v, false, b[j].v, (short)0, acc[j], false, false);
        }
        if (kt + 1 < numKt) a = a_next;
    }

    const int rbase = (lane < 16) ? 0 : 8;
#pragma unroll
    for (int j = 0; j < 8; ++j) {
        int col = ncol0 + j * 16 + (lane & 15);
        float bv = bias ? bias[col] : 0.0f;
#pragma unroll
        for (int r = 0; r < 8; ++r) {
            float v = acc[j][r] + bv;
            if (fuse_relu) v = fmaxf(v, 0.0f);
            C[(size_t)(mrow + rbase + r) * M + col] = v;
        }
    }
}

// ---------------- aggregation ----------------
__global__ void k_init_bias(float* __restrict__ agg, const float* __restrict__ b, long long nd4, int d) {
    long long i = (long long)blockIdx.x * blockDim.x + threadIdx.x;
    if (i >= nd4) return;
    int c = (int)((i * 4) % d);
    float4 bv = *(const float4*)(b + c);
    *(float4*)(agg + i * 4) = bv;
}

__global__ void k_aggregate(const float* __restrict__ T, float* __restrict__ AGG,
                            const int* __restrict__ src, const int* __restrict__ dst,
                            const float* __restrict__ dinv, int E_, int N_, int d)
{
    long long tid = (long long)blockIdx.x * blockDim.x + threadIdx.x;
    int chunks = d >> 2;
    long long total = (long long)(E_ + N_) * chunks;
    if (tid >= total) return;
    int e = (int)(tid / chunks);
    int c = (int)(tid % chunks) * 4;
    int s, dn; float w;
    if (e < E_) { s = src[e]; dn = dst[e]; w = dinv[s] * dinv[dn]; }
    else        { s = dn = e - E_; float di = dinv[s]; w = di * di; }
    float4 v = *(const float4*)(T + (size_t)s * d + c);
    float* o = AGG + (size_t)dn * d + c;
    atomicAdd(o + 0, v.x * w);
    atomicAdd(o + 1, v.y * w);
    atomicAdd(o + 2, v.z * w);
    atomicAdd(o + 3, v.w * w);
}

// relu -> eval BatchNorm; writes f32 and bf16 copies
__global__ void k_bn_relu(const float* __restrict__ AGG, float* __restrict__ Hf,
                          unsigned short* __restrict__ Hb,
                          const float* __restrict__ g, const float* __restrict__ be,
                          const float* __restrict__ m, const float* __restrict__ v,
                          long long nd4, int d)
{
    long long i = (long long)blockIdx.x * blockDim.x + threadIdx.x;
    if (i >= nd4) return;
    long long base = i * 4;
    int c = (int)(base % d);
    float4 x = *(const float4*)(AGG + base);
    float o0, o1, o2, o3;
    {
        float s0 = g[c+0] * rsqrtf(v[c+0] + 1e-5f);
        float s1 = g[c+1] * rsqrtf(v[c+1] + 1e-5f);
        float s2 = g[c+2] * rsqrtf(v[c+2] + 1e-5f);
        float s3 = g[c+3] * rsqrtf(v[c+3] + 1e-5f);
        o0 = s0 * (fmaxf(x.x, 0.f) - m[c+0]) + be[c+0];
        o1 = s1 * (fmaxf(x.y, 0.f) - m[c+1]) + be[c+1];
        o2 = s2 * (fmaxf(x.z, 0.f) - m[c+2]) + be[c+2];
        o3 = s3 * (fmaxf(x.w, 0.f) - m[c+3]) + be[c+3];
    }
    *(float4*)(Hf + base) = make_float4(o0, o1, o2, o3);
    ushort4 ob; ob.x = f2bf(o0); ob.y = f2bf(o1); ob.z = f2bf(o2); ob.w = f2bf(o3);
    *(ushort4*)(Hb + base) = ob;
}

// ---------------- pooling ----------------
__global__ void k_zero(float* p, long long n) {
    long long i = (long long)blockIdx.x * blockDim.x + threadIdx.x;
    if (i < n) p[i] = 0.0f;
}
__global__ void k_pool_scatter(const float* __restrict__ H, const int* __restrict__ batch,
                               float* __restrict__ sums, float* __restrict__ cnt, int n, int d)
{
    long long tid = (long long)blockIdx.x * blockDim.x + threadIdx.x;
    int chunks = d >> 2;
    long long total = (long long)n * chunks;
    if (tid >= total) return;
    int node = (int)(tid / chunks);
    int c = (int)(tid % chunks) * 4;
    int gidx = batch[node];
    float4 v = *(const float4*)(H + (size_t)node * d + c);
    float* o = sums + (size_t)gidx * d + c;
    atomicAdd(o + 0, v.x); atomicAdd(o + 1, v.y);
    atomicAdd(o + 2, v.z); atomicAdd(o + 3, v.w);
    if (c == 0) atomicAdd(&cnt[gidx], 1.0f);
}
__global__ void k_pool_finalize(const float* __restrict__ sums, const float* __restrict__ cnt,
                                unsigned short* __restrict__ out_bf, int gcount, int d)
{
    int i = blockIdx.x * blockDim.x + threadIdx.x;
    if (i >= gcount * d) return;
    int gidx = i / d;
    out_bf[i] = f2bf(sums[i] / fmaxf(cnt[gidx], 1.0f));
}

// ---------------- host orchestration ----------------
static inline long long cdiv(long long a, long long b) { return (a + b - 1) / b; }

extern "C" void kernel_launch(void* const* d_in, const int* in_sizes, int n_in,
                              void* d_out, int out_size, void* d_ws, size_t ws_size,
                              hipStream_t stream) {
    const float* x     = (const float*)d_in[0];
    const int*   ei    = (const int*)d_in[1];
    const int*   batch = (const int*)d_in[2];
    const float* W1 = (const float*)d_in[3];  const float* b1 = (const float*)d_in[4];
    const float* W2 = (const float*)d_in[5];  const float* b2 = (const float*)d_in[6];
    const float* W3 = (const float*)d_in[7];  const float* b3 = (const float*)d_in[8];
    const float* g1 = (const float*)d_in[9];  const float* be1 = (const float*)d_in[10];
    const float* m1 = (const float*)d_in[11]; const float* v1  = (const float*)d_in[12];
    const float* g2 = (const float*)d_in[13]; const float* be2 = (const float*)d_in[14];
    const float* m2 = (const float*)d_in[15]; const float* v2  = (const float*)d_in[16];
    const float* g3 = (const float*)d_in[17]; const float* be3 = (const float*)d_in[18];
    const float* m3 = (const float*)d_in[19]; const float* v3  = (const float*)d_in[20];
    const float* Wf1 = (const float*)d_in[21]; const float* bf1 = (const float*)d_in[22];
    const float* Wf2 = (const float*)d_in[23]; const float* bf2 = (const float*)d_in[24];

    const int D_H  = in_sizes[4];            // 512
    const int D_O  = in_sizes[8];            // 256
    const int D_IN = in_sizes[3] / D_H;      // 128
    const int N    = in_sizes[0] / D_IN;     // 131072
    const int E    = in_sizes[1] / 2;        // 524288
    const int G    = out_size / D_O;         // 4096
    const int* src = ei;
    const int* dst = ei + E;

    // ---- workspace carving (256B aligned) ----
    char* ws = (char*)d_ws;
    size_t off = 0;
    auto carve = [&](size_t bytes) { char* p = ws + off; off = (off + bytes + 255) & ~(size_t)255; return p; };
    float*          DINV = (float*)         carve((size_t)N * 4);
    unsigned short* ABUF = (unsigned short*)carve((size_t)N * D_H * 2);   // bf16 layer input
    float*          TBUF = (float*)         carve((size_t)N * D_H * 4);   // GEMM out / f32 features
    float*          AGG  = (float*)         carve((size_t)N * D_H * 4);   // aggregation accumulator
    unsigned short* WP1  = (unsigned short*)carve((size_t)D_IN * D_H * 2);
    unsigned short* WP2  = (unsigned short*)carve((size_t)D_H * D_H * 2);
    unsigned short* WP3  = (unsigned short*)carve((size_t)D_H * D_O * 2);
    unsigned short* WPF1 = (unsigned short*)carve((size_t)D_O * D_O * 2);
    unsigned short* WPF2 = (unsigned short*)carve((size_t)D_O * D_O * 2);
    float*          POOLS= (float*)         carve((size_t)G * D_O * 4);
    float*          CNT  = (float*)         carve((size_t)G * 4);
    unsigned short* PBF  = (unsigned short*)carve((size_t)G * D_O * 2);
    float*          HT   = (float*)         carve((size_t)G * D_O * 4);
    unsigned short* HTB  = (unsigned short*)carve((size_t)G * D_O * 2);
    (void)ws_size; (void)n_in;

    const int TB = 256;

    // ---- degree / symmetric norm ----
    k_init_deg<<<cdiv(N, TB), TB, 0, stream>>>(DINV, N);
    k_add_deg <<<cdiv(E, TB), TB, 0, stream>>>(dst, DINV, E);
    k_dinv    <<<cdiv(N, TB), TB, 0, stream>>>(DINV, N);

    // ---- pack all weights into WMMA B-fragment layout ----
    k_pack_w<<<cdiv((long long)D_IN * D_H, TB), TB, 0, stream>>>(W1,  WP1,  D_IN, D_H);
    k_pack_w<<<cdiv((long long)D_H  * D_H, TB), TB, 0, stream>>>(W2,  WP2,  D_H,  D_H);
    k_pack_w<<<cdiv((long long)D_H  * D_O, TB), TB, 0, stream>>>(W3,  WP3,  D_H,  D_O);
    k_pack_w<<<cdiv((long long)D_O  * D_O, TB), TB, 0, stream>>>(Wf1, WPF1, D_O,  D_O);
    k_pack_w<<<cdiv((long long)D_O  * D_O, TB), TB, 0, stream>>>(Wf2, WPF2, D_O,  D_O);

    // ---- input features -> bf16 ----
    k_f32_to_bf16<<<cdiv((long long)N * D_IN / 4, TB), TB, 0, stream>>>(x, ABUF, (long long)N * D_IN / 4);

    struct Layer { const unsigned short* Wp; const float *b, *g, *be, *m, *v; int K, M; };
    Layer L[3] = {
        { WP1, b1, g1, be1, m1, v1, D_IN, D_H },
        { WP2, b2, g2, be2, m2, v2, D_H,  D_H },
        { WP3, b3, g3, be3, m3, v3, D_H,  D_O },
    };

    for (int li = 0; li < 3; ++li) {
        const int K = L[li].K, M = L[li].M;
        dim3 gg(N / 128, M / 128);
        k_gemm_wmma<<<gg, TB, 0, stream>>>(ABUF, L[li].Wp, TBUF, nullptr, K, M, 0);
        long long nd4 = (long long)N * M / 4;
        k_init_bias<<<cdiv(nd4, TB), TB, 0, stream>>>(AGG, L[li].b, nd4, M);
        long long tot = (long long)(E + N) * (M / 4);
        k_aggregate<<<cdiv(tot, TB), TB, 0, stream>>>(TBUF, AGG, src, dst, DINV, E, N, M);
        k_bn_relu<<<cdiv(nd4, TB), TB, 0, stream>>>(AGG, TBUF, ABUF,
                                                    L[li].g, L[li].be, L[li].m, L[li].v, nd4, M);
    }

    // ---- global mean pool (TBUF holds f32 h3 [N x D_O]) ----
    k_zero<<<cdiv((long long)G * D_O, TB), TB, 0, stream>>>(POOLS, (long long)G * D_O);
    k_zero<<<cdiv((long long)G, TB), TB, 0, stream>>>(CNT, (long long)G);
    k_pool_scatter<<<cdiv((long long)N * (D_O / 4), TB), TB, 0, stream>>>(TBUF, batch, POOLS, CNT, N, D_O);
    k_pool_finalize<<<cdiv((long long)G * D_O, TB), TB, 0, stream>>>(POOLS, CNT, PBF, G, D_O);

    // ---- MLP head: relu(pooled @ Wf1 + bf1) @ Wf2 + bf2 ----
    {
        dim3 gg(G / 128, D_O / 128);
        k_gemm_wmma<<<gg, TB, 0, stream>>>(PBF, WPF1, HT, bf1, D_O, D_O, 1);
        k_f32_to_bf16<<<cdiv((long long)G * D_O / 4, TB), TB, 0, stream>>>(HT, HTB, (long long)G * D_O / 4);
        k_gemm_wmma<<<gg, TB, 0, stream>>>(HTB, WPF2, (float*)d_out, bf2, D_O, D_O, 0);
    }
}
